// QuantizedMLP_163208757492
// MI455X (gfx1250) — 8-bit matrix operand order repaired and checked by reading, compiled, not run on hardware
//
#include <hip/hip_runtime.h>
#include <hip/hip_bf16.h>

typedef int  v8i  __attribute__((ext_vector_type(8)));
typedef unsigned int uint32;

#define QMAX      127.0f
#define D_IN      512
#define HID       64
#define C_OUT     2

#define OFF_W1Q   0
#define OFF_W2Q   32768
#define OFF_W3Q   36864
#define OFF_B1I   36992
#define OFF_B2I   37248
#define OFF_B3I   37504
#define OFF_SW1   37512
#define OFF_SW2   37516
#define OFF_SW3   37520
#define OFF_MAX1  37524
#define OFF_MAX2  37528
#define OFF_F1    37532
#define OFF_S1    37536
#define OFF_F2    37540
#define OFF_OUTS  37544
#define OFF_R1    40960

__device__ __forceinline__ float po2_scale_dev(float maxabs) {
    return exp2f(ceilf(log2f(maxabs / QMAX + 1e-12f)));
}

__device__ __forceinline__ int q8s_i(float v) {
    return (int)__builtin_amdgcn_fmed3f(rintf(v * QMAX), -QMAX, QMAX);
}
__device__ __forceinline__ int q8u_i(int r, float f) {
    return (int)__builtin_amdgcn_fmed3f(rintf((float)r * f), 0.0f, QMAX);
}
__device__ __forceinline__ int pack4b(int q0, int q1, int q2, int q3) {
    uint32 t01 = __builtin_amdgcn_perm((uint32)q1, (uint32)q0, 0x0c0c0400u);
    uint32 t23 = __builtin_amdgcn_perm((uint32)q3, (uint32)q2, 0x0c0c0400u);
    return (int)__builtin_amdgcn_perm(t23, t01, 0x05040100u);
}

__global__ __launch_bounds__(256) void prep_kernel(
    const float* __restrict__ W1, const float* __restrict__ b1,
    const float* __restrict__ W2, const float* __restrict__ W3,
    char* __restrict__ ws)
{
    __shared__ float red[256];
    const int tid = threadIdx.x;
    if (tid == 0) { *(int*)(ws + OFF_MAX1) = 0; *(int*)(ws + OFF_MAX2) = 0; }

    float m1 = 0.f, m2 = 0.f, m3 = 0.f;
    for (int i = tid; i < HID * D_IN; i += 256) m1 = fmaxf(m1, fabsf(W1[i]));
    for (int i = tid; i < HID * HID;  i += 256) m2 = fmaxf(m2, fabsf(W2[i]));
    for (int i = tid; i < C_OUT * HID; i += 256) m3 = fmaxf(m3, fabsf(W3[i]));

    float M[3]; float loc[3] = {m1, m2, m3};
    for (int a = 0; a < 3; ++a) {
        red[tid] = loc[a]; __syncthreads();
        for (int off = 128; off > 0; off >>= 1) {
            if (tid < off) red[tid] = fmaxf(red[tid], red[tid + off]);
            __syncthreads();
        }
        M[a] = red[0]; __syncthreads();
    }
    const float sw1 = po2_scale_dev(M[0]);
    const float sw2 = po2_scale_dev(M[1]);
    const float sw3 = po2_scale_dev(M[2]);
    if (tid == 0) {
        *(float*)(ws + OFF_SW1) = sw1;
        *(float*)(ws + OFF_SW2) = sw2;
        *(float*)(ws + OFF_SW3) = sw3;
    }
    signed char* w1q = (signed char*)(ws + OFF_W1Q);
    signed char* w2q = (signed char*)(ws + OFF_W2Q);
    signed char* w3q = (signed char*)(ws + OFF_W3Q);
    for (int i = tid; i < HID * D_IN; i += 256)
        w1q[i] = (signed char)(int)__builtin_amdgcn_fmed3f(rintf(W1[i] / sw1), -QMAX, QMAX);
    for (int i = tid; i < HID * HID; i += 256)
        w2q[i] = (signed char)(int)__builtin_amdgcn_fmed3f(rintf(W2[i] / sw2), -QMAX, QMAX);
    for (int i = tid; i < C_OUT * HID; i += 256)
        w3q[i] = (signed char)(int)__builtin_amdgcn_fmed3f(rintf(W3[i] / sw3), -QMAX, QMAX);
    if (tid < HID) {
        ((int*)(ws + OFF_B1I))[tid] = (int)rintf(b1[tid] * QMAX / sw1);
    }
}

#define W1_PITCH 132

__global__ __launch_bounds__(256) void gemm1_kernel(
    const float* __restrict__ x, char* __restrict__ ws, int* __restrict__ r1out)
{
    __shared__ uint32 ldsw[HID * W1_PITCH];
    const uint32* w1q32 = (const uint32*)(ws + OFF_W1Q);
    const int tid = threadIdx.x;
    for (int i = tid; i < HID * 128; i += 256) {
        int row = i >> 7, col = i & 127;
        ldsw[row * W1_PITCH + col] = w1q32[i];
    }
    __syncthreads();

    const int lane  = tid & 31;
    const int wave  = tid >> 5;
    const int lhalf = lane >> 4;
    const int l15   = lane & 15;

    const size_t rowA = (size_t)blockIdx.x * 128 + wave * 16 + l15;
    const float* xrow = x + rowA * D_IN;
    const int*   b1i  = (const int*)(ws + OFF_B1I);

    v8i acc[4];
#pragma unroll
    for (int t = 0; t < 4; ++t)
#pragma unroll
        for (int r = 0; r < 8; ++r) acc[t][r] = 0;

    const int kselA = lhalf * 16;
    for (int kb = 0; kb < D_IN; kb += 64) {
        v8i A;
#pragma unroll
        for (int v = 0; v < 8; ++v) {
            const int k0 = kb + ((v >> 2) << 5)
                              + ((v & 3) << 2) + kselA;
            const float4 f = *(const float4*)(xrow + k0);
            A[v] = pack4b(q8s_i(f.x), q8s_i(f.y), q8s_i(f.z), q8s_i(f.w));
        }
        const int kd = (kb >> 2) + lhalf * 4;
#pragma unroll
        for (int t = 0; t < 4; ++t) {
            const int n = t * 16 + l15;
            const uint4 b01 = *(const uint4*)&ldsw[n * W1_PITCH + kd];
            const uint4 b23 = *(const uint4*)&ldsw[n * W1_PITCH + kd + 8];
            v8i Bf;
            Bf[0] = (int)b01.x; Bf[1] = (int)b01.y; Bf[2] = (int)b01.z; Bf[3] = (int)b01.w;
            Bf[4] = (int)b23.x; Bf[5] = (int)b23.y; Bf[6] = (int)b23.z; Bf[7] = (int)b23.w;
            acc[t] = __builtin_amdgcn_wmma_i32_16x16x64_iu8(
                true, A, true, Bf, acc[t], false, false);
        }
    }

    int localmax = 0;
    const size_t outRowBase = (size_t)blockIdx.x * 128 + wave * 16 + lhalf * 8;
#pragma unroll
    for (int t = 0; t < 4; ++t) {
        const int n    = t * 16 + l15;
        const int bias = b1i[n];
#pragma unroll
        for (int r = 0; r < 8; ++r) {
            int h = acc[t][r] + bias;
            h = h > 0 ? h : 0;
            r1out[(outRowBase + r) * HID + n] = h;
            localmax = h > localmax ? h : localmax;
        }
    }
    for (int off = 16; off > 0; off >>= 1) {
        int o = __shfl_xor(localmax, off, 32);
        localmax = o > localmax ? o : localmax;
    }
    if (lane == 0) atomicMax((int*)(ws + OFF_MAX1), localmax);
}

__global__ void finalize1_kernel(const float* __restrict__ b2, char* __restrict__ ws)
{
    const int tid = threadIdx.x;
    const float sw1 = *(const float*)(ws + OFF_SW1);
    const float sw2 = *(const float*)(ws + OFF_SW2);
    const int   m   = *(const int*)(ws + OFF_MAX1);
    const float rmax = (float)m * (sw1 / QMAX);
    const float s1   = po2_scale_dev(rmax);
    if (tid < HID) ((int*)(ws + OFF_B2I))[tid] = (int)rintf(b2[tid] / (s1 * sw2));
    if (tid == 0) {
        *(float*)(ws + OFF_F1) = (sw1 / QMAX) / s1;
        *(float*)(ws + OFF_S1) = s1;
    }
}

#define W2_PITCH 20

__global__ __launch_bounds__(256) void gemm2_kernel(
    char* __restrict__ ws, const int* __restrict__ r1in, int* __restrict__ r2out)
{
    __shared__ uint32 ldsw[HID * W2_PITCH];
    const uint32* w2q32 = (const uint32*)(ws + OFF_W2Q);
    const int tid = threadIdx.x;
    for (int i = tid; i < HID * 16; i += 256) {
        int row = i >> 4, col = i & 15;
        ldsw[row * W2_PITCH + col] = w2q32[i];
    }
    const float f1 = *(const float*)(ws + OFF_F1);
    __syncthreads();

    const int lane  = tid & 31;
    const int wave  = tid >> 5;
    const int lhalf = lane >> 4;
    const int l15   = lane & 15;

    const size_t rowA = (size_t)blockIdx.x * 128 + wave * 16 + l15;
    const int* arow   = r1in + rowA * HID;
    const int* b2i    = (const int*)(ws + OFF_B2I);

    const int kselA = lhalf * 16;
    v8i A;
#pragma unroll
    for (int v = 0; v < 8; ++v) {
        const int k0 = ((v >> 2) << 5) + ((v & 3) << 2) + kselA;
        const int4 iv = *(const int4*)(arow + k0);
        A[v] = pack4b(q8u_i(iv.x, f1), q8u_i(iv.y, f1),
                      q8u_i(iv.z, f1), q8u_i(iv.w, f1));
    }

    v8i acc[4];
#pragma unroll
    for (int t = 0; t < 4; ++t)
#pragma unroll
        for (int r = 0; r < 8; ++r) acc[t][r] = 0;

    const int kd = lhalf * 4;
#pragma unroll
    for (int t = 0; t < 4; ++t) {
        const int n = t * 16 + l15;
        const uint4 b01 = *(const uint4*)&ldsw[n * W2_PITCH + kd];
        const uint4 b23 = *(const uint4*)&ldsw[n * W2_PITCH + kd + 8];
        v8i Bf;
        Bf[0] = (int)b01.x; Bf[1] = (int)b01.y; Bf[2] = (int)b01.z; Bf[3] = (int)b01.w;
        Bf[4] = (int)b23.x; Bf[5] = (int)b23.y; Bf[6] = (int)b23.z; Bf[7] = (int)b23.w;
        acc[t] = __builtin_amdgcn_wmma_i32_16x16x64_iu8(
            true, A, true, Bf, acc[t], false, false);
    }

    int localmax = 0;
    const size_t outRowBase = (size_t)blockIdx.x * 128 + wave * 16 + lhalf * 8;
#pragma unroll
    for (int t = 0; t < 4; ++t) {
        const int n    = t * 16 + l15;
        const int bias = b2i[n];
#pragma unroll
        for (int r = 0; r < 8; ++r) {
            int h = acc[t][r] + bias;
            h = h > 0 ? h : 0;
            r2out[(outRowBase + r) * HID + n] = h;
            localmax = h > localmax ? h : localmax;
        }
    }
    for (int off = 16; off > 0; off >>= 1) {
        int o = __shfl_xor(localmax, off, 32);
        localmax = o > localmax ? o : localmax;
    }
    if (lane == 0) atomicMax((int*)(ws + OFF_MAX2), localmax);
}

__global__ void finalize2_kernel(const float* __restrict__ b3, char* __restrict__ ws)
{
    const int tid = threadIdx.x;
    const float s1  = *(const float*)(ws + OFF_S1);
    const float sw2 = *(const float*)(ws + OFF_SW2);
    const float sw3 = *(const float*)(ws + OFF_SW3);
    const int   m   = *(const int*)(ws + OFF_MAX2);
    const float rmax = (float)m * (s1 * sw2);
    const float s2   = po2_scale_dev(rmax);
    if (tid < C_OUT) ((int*)(ws + OFF_B3I))[tid] = (int)rintf(b3[tid] / (s2 * sw3));
    if (tid == 0) {
        *(float*)(ws + OFF_F2)   = (s1 * sw2) / s2;
        *(float*)(ws + OFF_OUTS) = s2 * sw3;
    }
}

__global__ __launch_bounds__(256) void gemm3_kernel(
    const char* __restrict__ ws, const int* __restrict__ r2in, float* __restrict__ out)
{
    const size_t row = (size_t)blockIdx.x * 256 + threadIdx.x;
    const int4* r2 = (const int4*)(r2in + row * HID);
    const signed char* w3q = (const signed char*)(ws + OFF_W3Q);
    const float f2   = *(const float*)(ws + OFF_F2);
    const float outs = *(const float*)(ws + OFF_OUTS);
    const int*  b3i  = (const int*)(ws + OFF_B3I);
    int s0 = b3i[0], s1 = b3i[1];
#pragma unroll
    for (int j = 0; j < 16; ++j) {
        const int4 v = r2[j];
        int q;
        q = q8u_i(v.x, f2); s0 += q * (int)w3q[j*4+0]; s1 += q * (int)w3q[64 + j*4+0];
        q = q8u_i(v.y, f2); s0 += q * (int)w3q[j*4+1]; s1 += q * (int)w3q[64 + j*4+1];
        q = q8u_i(v.z, f2); s0 += q * (int)w3q[j*4+2]; s1 += q * (int)w3q[64 + j*4+2];
        q = q8u_i(v.w, f2); s0 += q * (int)w3q[j*4+3]; s1 += q * (int)w3q[64 + j*4+3];
    }
    out[row * 2 + 0] = (float)s0 * outs;
    out[row * 2 + 1] = (float)s1 * outs;
}

extern "C" void kernel_launch(void* const* d_in, const int* in_sizes, int n_in,
                              void* d_out, int out_size, void* d_ws, size_t ws_size,
                              hipStream_t stream) {
    const float* x  = (const float*)d_in[0];
    const float* W1 = (const float*)d_in[1];
    const float* b1 = (const float*)d_in[2];
    const float* W2 = (const float*)d_in[3];
    const float* b2 = (const float*)d_in[4];
    const float* W3 = (const float*)d_in[5];
    const float* b3 = (const float*)d_in[6];
    float* out = (float*)d_out;

    const int B = in_sizes[0] / D_IN;
    char* ws = (char*)d_ws;
    int* r1 = (int*)(ws + OFF_R1);
    int* r2 = (int*)(ws + OFF_R1 + (size_t)B * HID * sizeof(int));

    prep_kernel<<<1, 256, 0, stream>>>(W1, b1, W2, W3, ws);
    gemm1_kernel<<<B / 128, 256, 0, stream>>>(x, ws, r1);
    finalize1_kernel<<<1, 64, 0, stream>>>(b2, ws);
    gemm2_kernel<<<B / 128, 256, 0, stream>>>(ws, r1, r2);
    finalize2_kernel<<<1, 32, 0, stream>>>(b3, ws);
    gemm3_kernel<<<B / 256, 256, 0, stream>>>(ws, r2, out);
}
